// GeometricEdgeEncoder_50448685859296
// MI455X (gfx1250) — matrix operand order repaired and checked by reading, not compiled, not run on hardware
//
#include <hip/hip_runtime.h>

typedef __attribute__((ext_vector_type(2)))  float    v2f;
typedef __attribute__((ext_vector_type(8)))  float    v8f;
typedef __attribute__((ext_vector_type(8)))  _Float16 v8h;
typedef __attribute__((ext_vector_type(16))) _Float16 v16h;

#define HID   64
#define OUTD  32
#define WAVES 8

static __device__ __forceinline__ v8f splat8(float x) {
    v8f v;
#pragma unroll
    for (int k = 0; k < 8; ++k) v[k] = x;
    return v;
}

__global__ __launch_bounds__(256) void edge_mlp_wmma(
    const float* __restrict__ pos, const float* __restrict__ W1,
    const float* __restrict__ b1,  const float* __restrict__ W2,
    const float* __restrict__ b2,  float* __restrict__ out, int N)
{
    // per-wave scratch: 16 edges x 64 hidden, f16, row-major [m][k]
    __shared__ alignas(16) _Float16 hs[WAVES][16 * HID];

    const int tid  = threadIdx.x;
    const int wave = tid >> 5;
    const int lane = tid & 31;
    const int lm   = lane & 15;
    const bool hiH = lane >= 16;   // which 16-lane half

    const int i = blockIdx.x;

    // node-i position (wave-uniform)
    const float pix = pos[3 * i + 0];
    const float piy = pos[3 * i + 1];
    const float piz = pos[3 * i + 2];

    // ---- GEMM1 B operands: W1 (7x64) padded to K=8, f32 B 4x16 layout ----
    // VGPR0: K = hi?2:0 ; VGPR1: K = hi?3:1   (second wmma: 4/6 and 5/7, K7 = pad 0)
    v2f   w1bA[4], w1bB[4];
    float bias1[4];
#pragma unroll
    for (int t = 0; t < 4; ++t) {
        const int col = t * 16 + lm;
        w1bA[t].x = W1[(hiH ? 2 : 0) * HID + col];
        w1bA[t].y = W1[(hiH ? 3 : 1) * HID + col];
        w1bB[t].x = W1[(hiH ? 6 : 4) * HID + col];
        w1bB[t].y = hiH ? 0.0f : W1[5 * HID + col];
        bias1[t]  = b1[col];
    }

    // ---- GEMM2 B operands: W2 (64x32) as f16, 16-bit B 32x16 layout ----
    // lane: n = nt*16+lm ; k = c*32 + (idx<8 ? (hi?8:0)+idx : 16+(hi?8:0)+idx-8), idx = 0..15
    v16h w2b[2][2];
    v8f  cinit[2];
#pragma unroll
    for (int nt = 0; nt < 2; ++nt) {
        const int n = nt * 16 + lm;
#pragma unroll
        for (int c = 0; c < 2; ++c) {
#pragma unroll
            for (int idx = 0; idx < 16; ++idx) {
                const int k = c * 32 + (idx < 8 ? (hiH ? 8 : 0) + idx : 16 + (hiH ? 8 : 0) + (idx - 8));
                w2b[nt][c][idx] = (_Float16)W2[k * OUTD + n];
            }
        }
        cinit[nt] = splat8(b2[n]);   // fold b2 into the accumulator init
    }

    const int jTiles = N >> 4;
    for (int jt = wave; jt < jTiles; jt += WAVES) {
        const int jbase = jt << 4;
        const int j     = jbase + lm;

        // ---- edge features for this lane's j ----
        const float dx = pix - pos[3 * j + 0];
        const float dy = piy - pos[3 * j + 1];
        const float dz = piz - pos[3 * j + 2];
        const float d2 = fmaf(dx, dx, fmaf(dy, dy, dz * dz));
        float dist = __builtin_amdgcn_sqrtf(d2);
        dist = fmaxf(dist, 1e-8f);
        const float rd = __builtin_amdgcn_rcpf(dist);
        const float ux = dx * rd, uy = dy * rd, uz = dz * rd;

        // A operands, f32 A 16x4 layout: K slice {dx,dy,dz,dist} then {ux,uy,uz,0}
        v2f aLo, aHi;
        aLo.x = hiH ? dz   : dx;
        aLo.y = hiH ? dist : dy;
        aHi.x = hiH ? uz   : ux;
        aHi.y = hiH ? 0.0f : uy;

        // ---- GEMM1 (8x v_wmma_f32_16x16x4_f32) + SiLU -> LDS f16 ----
#pragma unroll
        for (int t = 0; t < 4; ++t) {
            v8f acc = splat8(bias1[t]);
            acc = __builtin_amdgcn_wmma_f32_16x16x4_f32(
                      false, aLo, false, w1bA[t], (short)0, acc, false, false);
            acc = __builtin_amdgcn_wmma_f32_16x16x4_f32(
                      false, aHi, false, w1bB[t], (short)0, acc, false, false);
#pragma unroll
            for (int r = 0; r < 8; ++r) {
                const float x = acc[r];
                const float e = __builtin_amdgcn_exp2f(-1.44269504f * x);
                const float s = x * __builtin_amdgcn_rcpf(1.0f + e);   // SiLU
                const int   m = r + (hiH ? 8 : 0);                     // C row
                hs[wave][m * HID + t * 16 + lm] = (_Float16)s;
            }
        }
        // order LDS store -> LDS load within the wave
        asm volatile("s_wait_dscnt 0" ::: "memory");

        // ---- repack to 16-bit A 16x32 layout (2x ds_load_b128 per chunk) ----
        const int koff = hiH ? 8 : 0;
        v16h a2[2];
#pragma unroll
        for (int c = 0; c < 2; ++c) {
            const v8h g0 = *(const v8h*)&hs[wave][lm * HID + c * 32 + koff];
            const v8h g1 = *(const v8h*)&hs[wave][lm * HID + c * 32 + 16 + koff];
            a2[c] = __builtin_shufflevector(g0, g1,
                        0, 1, 2, 3, 4, 5, 6, 7, 8, 9, 10, 11, 12, 13, 14, 15);
        }

        // ---- GEMM2 (4x v_wmma_f32_16x16x32_f16) + streaming stores ----
#pragma unroll
        for (int nt = 0; nt < 2; ++nt) {
            v8f c = cinit[nt];
            c = __builtin_amdgcn_wmma_f32_16x16x32_f16(
                    false, a2[0], false, w2b[nt][0], (short)0, c, false, false);
            c = __builtin_amdgcn_wmma_f32_16x16x32_f16(
                    false, a2[1], false, w2b[nt][1], (short)0, c, false, false);
#pragma unroll
            for (int r = 0; r < 8; ++r) {
                const int m = r + (hiH ? 8 : 0);
                float* p = &out[((size_t)i * N + (jbase + m)) * OUTD + nt * 16 + lm];
                __builtin_nontemporal_store(c[r], p);   // 512 MB stream: bypass L2 retention
            }
        }
    }
}

extern "C" void kernel_launch(void* const* d_in, const int* in_sizes, int n_in,
                              void* d_out, int out_size, void* d_ws, size_t ws_size,
                              hipStream_t stream) {
    const float* pos = (const float*)d_in[0];
    const float* W1  = (const float*)d_in[1];
    const float* b1  = (const float*)d_in[2];
    const float* W2  = (const float*)d_in[3];
    const float* b2  = (const float*)d_in[4];
    float* out = (float*)d_out;

    const int N = in_sizes[0] / 3;           // positions is (N, 3)
    dim3 grid(N), block(WAVES * 32);
    hipLaunchKernelGGL(edge_mlp_wmma, grid, block, 0, stream,
                       pos, W1, b1, W2, b2, out, N);
}
